// RDEC_38766374813726
// MI455X (gfx1250) — compile-verified, matrix operand order checked by reading, not run on hardware
//
#include <hip/hip_runtime.h>

// ---------------------------------------------------------------------------
// DEC soft-assignment: q = normalize_rows( 1 / (1 + ||x - c||^2) )
//   cross GEMM done with split-bf16 (hi+lo) on v_wmma_f32_16x16x32_bf16.
// N=65536 rows, D=256 features, K=1024 centroids.
// ---------------------------------------------------------------------------

typedef __attribute__((ext_vector_type(16))) __bf16 v16bf;
typedef __attribute__((ext_vector_type(8)))  float  v8f;
typedef __attribute__((ext_vector_type(4)))  float  f32x4;

#define N_TOTAL 65536
#define DDIM    256
#define KCENT   1024
#define M_TILE  16     // rows per block
#define NWAVES  8      // 256 threads
#define TPW     8      // 16-col WMMA tiles per wave (8*8*16 = 1024 cols)
#define KC      32     // K-slice per bf16 WMMA

// ---------------------------------------------------------------------------
// Prep: centroids f32 -> bf16 hi/lo split + per-centroid squared norms.
// One block per centroid (256 threads == D).
// ---------------------------------------------------------------------------
__global__ __launch_bounds__(256) void dec_prep(
    const float* __restrict__ cent,
    __bf16* __restrict__ chi, __bf16* __restrict__ clo,
    float* __restrict__ csq) {
  const int k = blockIdx.x;
  const int d = threadIdx.x;
  const float v = cent[(size_t)k * DDIM + d];
  const __bf16 h = (__bf16)v;
  const float  rem = v - (float)h;
  chi[(size_t)k * DDIM + d] = h;
  clo[(size_t)k * DDIM + d] = (__bf16)rem;

  float s = v * v;
  #pragma unroll
  for (int m = 1; m < 32; m <<= 1) s += __shfl_xor(s, m, 32);
  __shared__ float wsum[NWAVES];
  const int lane = threadIdx.x & 31, wv = threadIdx.x >> 5;
  if (lane == 0) wsum[wv] = s;
  __syncthreads();
  if (threadIdx.x == 0) {
    float t = 0.f;
    #pragma unroll
    for (int i = 0; i < NWAVES; ++i) t += wsum[i];
    csq[k] = t;
  }
}

// ---------------------------------------------------------------------------
// Main: block = 16 rows x 1024 centroids (full row -> single-pass normalize).
// ---------------------------------------------------------------------------
__global__ __launch_bounds__(256) void dec_main(
    const float*  __restrict__ x,
    const __bf16* __restrict__ chi,
    const __bf16* __restrict__ clo,
    const float*  __restrict__ csq,
    float*        __restrict__ out) {
  __shared__ float sxsq[M_TILE];   // ||x_row||^2
  __shared__ float srow[M_TILE];   // row sums of numerator

  const int tid  = threadIdx.x;
  const int lane = tid & 31;
  const int wave = tid >> 5;
  const int row0 = blockIdx.x * M_TILE;

  // --- ||x||^2 for the 16 rows of this block (16 threads per row) ---------
  {
    const int r = tid >> 4;        // 0..15 row
    const int c = tid & 15;        // 16-float segment per thread
    const float* p = x + (size_t)(row0 + r) * DDIM + c * 16;
    float s = 0.f;
    #pragma unroll
    for (int i = 0; i < 16; i += 4) {
      f32x4 v = *(const f32x4*)(p + i);
      s += v.x * v.x + v.y * v.y + v.z * v.z + v.w * v.w;
    }
    #pragma unroll
    for (int m = 1; m < 16; m <<= 1) s += __shfl_xor(s, m, 32);
    if ((lane & 15) == 0) sxsq[r] = s;
    if (tid < M_TILE) srow[tid] = 0.f;
  }
  __syncthreads();

  // --- WMMA fragment coordinates (16-bit A/B layouts, wave32) -------------
  const int mrow = lane & 15;      // A: M index of this lane
  const int g    = lane >> 4;      // K half selector (0: K 0-15, 1: K 16-31)
  const int ncol = lane & 15;      // B: N index of this lane

  v8f acc[TPW];
  #pragma unroll
  for (int t = 0; t < TPW; ++t) acc[t] = (v8f)(0.f);

  const float* arow = x + (size_t)(row0 + mrow) * DDIM;

  for (int kc = 0; kc < DDIM; kc += KC) {
    // A fragment: 16 consecutive f32 of this lane's row, split hi/lo bf16.
    v16bf ahi, alo;
    {
      const float* ap = arow + kc + g * 16;
      #pragma unroll
      for (int i = 0; i < 16; i += 4) {
        f32x4 v = *(const f32x4*)(ap + i);
        float f0 = v.x, f1 = v.y, f2 = v.z, f3 = v.w;
        __bf16 h0 = (__bf16)f0, h1 = (__bf16)f1, h2 = (__bf16)f2, h3 = (__bf16)f3;
        ahi[i + 0] = h0; alo[i + 0] = (__bf16)(f0 - (float)h0);
        ahi[i + 1] = h1; alo[i + 1] = (__bf16)(f1 - (float)h1);
        ahi[i + 2] = h2; alo[i + 2] = (__bf16)(f2 - (float)h2);
        ahi[i + 3] = h3; alo[i + 3] = (__bf16)(f3 - (float)h3);
      }
    }
    #pragma unroll
    for (int t = 0; t < TPW; ++t) {
      const int col = wave * (TPW * 16) + t * 16 + ncol;
      const size_t boff = (size_t)col * DDIM + kc + g * 16;
      v16bf bhi = *(const v16bf*)(chi + boff);
      v16bf blo = *(const v16bf*)(clo + boff);
      // cross ~= hi*hi + lo*hi + hi*lo   (split-f32 emulation on bf16 WMMA)
      acc[t] = __builtin_amdgcn_wmma_f32_16x16x32_bf16(
          false, ahi, false, bhi, (short)0, acc[t], false, false);
      acc[t] = __builtin_amdgcn_wmma_f32_16x16x32_bf16(
          false, alo, false, bhi, (short)0, acc[t], false, false);
      acc[t] = __builtin_amdgcn_wmma_f32_16x16x32_bf16(
          false, ahi, false, blo, (short)0, acc[t], false, false);
    }
  }

  // --- Epilogue: numerator + partial row sums ------------------------------
  // C/D layout: VGPR r, lane l -> M = r + 8*(l>>4), N = (l&15) + tile base.
  float partial[8];
  #pragma unroll
  for (int r = 0; r < 8; ++r) partial[r] = 0.f;

  #pragma unroll
  for (int t = 0; t < TPW; ++t) {
    const int col = wave * (TPW * 16) + t * 16 + ncol;
    const float cs = csq[col];
    #pragma unroll
    for (int r = 0; r < 8; ++r) {
      float ns = sxsq[r + 8 * g] + cs - 2.f * acc[t][r];
      ns = fmaxf(ns, 0.f);
      const float num = __builtin_amdgcn_rcpf(1.f + ns);
      acc[t][r] = num;
      partial[r] += num;
    }
  }

  // 16-lane reduction (all N columns of one row live in one lane group).
  #pragma unroll
  for (int r = 0; r < 8; ++r) {
    float s = partial[r];
    s += __shfl_xor(s, 1, 32);
    s += __shfl_xor(s, 2, 32);
    s += __shfl_xor(s, 4, 32);
    s += __shfl_xor(s, 8, 32);
    if (ncol == 0) atomicAdd(&srow[r + 8 * g], s);   // ds_add_f32
  }
  __syncthreads();

  // --- Normalize + store ---------------------------------------------------
  #pragma unroll
  for (int r = 0; r < 8; ++r) {
    const int row = r + 8 * g;
    const float inv = __builtin_amdgcn_rcpf(srow[row]);
    #pragma unroll
    for (int t = 0; t < TPW; ++t) {
      const int col = wave * (TPW * 16) + t * 16 + ncol;
      out[(size_t)(row0 + row) * KCENT + col] = acc[t][r] * inv;
    }
  }
}

// ---------------------------------------------------------------------------
extern "C" void kernel_launch(void* const* d_in, const int* in_sizes, int n_in,
                              void* d_out, int out_size, void* d_ws, size_t ws_size,
                              hipStream_t stream) {
  (void)in_sizes; (void)n_in; (void)out_size; (void)ws_size;
  const float* x    = (const float*)d_in[0];
  const float* cent = (const float*)d_in[1];
  float*       out  = (float*)d_out;

  char* ws = (char*)d_ws;
  __bf16* chi = (__bf16*)ws;                                          // 512 KB
  __bf16* clo = (__bf16*)(ws + (size_t)KCENT * DDIM * sizeof(__bf16)); // 512 KB
  float*  csq = (float*)(ws + 2 * (size_t)KCENT * DDIM * sizeof(__bf16)); // 4 KB

  dec_prep<<<KCENT, 256, 0, stream>>>(cent, chi, clo, csq);
  dec_main<<<N_TOTAL / M_TILE, 256, 0, stream>>>(x, chi, clo, csq, out);
}
